// GraphAttentionLayer_21646635172724
// MI455X (gfx1250) — matrix operand order repaired and checked by reading, not compiled, not run on hardware
//
#include <hip/hip_runtime.h>
#include <hip/hip_bf16.h>

typedef __attribute__((ext_vector_type(16))) __bf16 v16bf;
typedef __attribute__((ext_vector_type(8)))  __bf16 v8bf;
typedef __attribute__((ext_vector_type(8)))  float  v8f;

#define N_NODES 8192
#define IN_F    512
#define OUT_F   256
#define NEG_SLOPE 0.2f
#define J_SPLIT 4          // j-chunks per i-tile (block count multiplier)
#define BROW    40         // staged B row stride in bf16 elems (64B data + 16B pad)

// ---------------------------------------------------------------------------
// Kernel 1: h = X @ W^T + bW   (bf16 WMMA, fp32 accumulate)
// ---------------------------------------------------------------------------
__global__ __launch_bounds__(128) void k_linear(const float* __restrict__ X,
                                                const float* __restrict__ W,
                                                const float* __restrict__ bW,
                                                float* __restrict__ h,
                                                __bf16* __restrict__ hT)
{
    const int tid  = threadIdx.x;
    const int lane = tid & 31;
    const int wid  = tid >> 5;
    const int tile = blockIdx.x * 4 + wid;     // 0..8191
    const int it   = tile >> 4;                // i-tile 0..511
    const int ft   = tile & 15;                // f-tile 0..15
    const int i0   = it * 16, f0 = ft * 16;
    const int m    = lane & 15;
    const int half = lane >> 4;

    const float* xrow = X + (i0 + m) * IN_F;
    const float* wrow = W + (f0 + m) * IN_F + half * 16;

    v8f acc = {};
    for (int k = 0; k < IN_F; k += 32) {
        v16bf a, b;
        #pragma unroll
        for (int v = 0; v < 8; ++v) {
            const int K0 = (half << 4) + (v << 1);
            float2 x2 = *(const float2*)(xrow + k + K0);
            a[2 * v]     = (__bf16)x2.x;
            a[2 * v + 1] = (__bf16)x2.y;
        }
        #pragma unroll
        for (int q = 0; q < 4; ++q) {
            float4 w4 = *(const float4*)(wrow + k + q * 4);
            b[4 * q]     = (__bf16)w4.x;
            b[4 * q + 1] = (__bf16)w4.y;
            b[4 * q + 2] = (__bf16)w4.z;
            b[4 * q + 3] = (__bf16)w4.w;
        }
        acc = __builtin_amdgcn_wmma_f32_16x16x32_bf16(false, a, false, b,
                                                      (short)0, acc, false, false);
    }

    const float bias = bW[f0 + m];
    #pragma unroll
    for (int r = 0; r < 8; ++r) {
        const int row = i0 + r + half * 8;
        const float val = acc[r] + bias;
        h[row * OUT_F + f0 + m] = val;
        hT[(f0 + m) * N_NODES + row] = (__bf16)val;
    }
}

// ---------------------------------------------------------------------------
// Kernel 2: src[i] = h[i]@a1 + ba, nbr[i] = h[i]@a2   (one wave per row)
// ---------------------------------------------------------------------------
__global__ __launch_bounds__(256) void k_srcnbr(const float* __restrict__ h,
                                                const float* __restrict__ a,
                                                const float* __restrict__ ba,
                                                float* __restrict__ src,
                                                float* __restrict__ nbr)
{
    const int lane = threadIdx.x & 31;
    const int row  = blockIdx.x * 8 + (threadIdx.x >> 5);
    const float* hr = h + row * OUT_F + lane * 8;
    const float* a1 = a + lane * 8;
    const float* a2 = a + OUT_F + lane * 8;
    float s1 = 0.f, s2 = 0.f;
    #pragma unroll
    for (int q = 0; q < 2; ++q) {
        float4 hv = *(const float4*)(hr + q * 4);
        float4 v1 = *(const float4*)(a1 + q * 4);
        float4 v2 = *(const float4*)(a2 + q * 4);
        s1 += hv.x * v1.x + hv.y * v1.y + hv.z * v1.z + hv.w * v1.w;
        s2 += hv.x * v2.x + hv.y * v2.y + hv.z * v2.z + hv.w * v2.w;
    }
    #pragma unroll
    for (int d = 16; d >= 1; d >>= 1) {
        s1 += __shfl_xor(s1, d, 32);
        s2 += __shfl_xor(s2, d, 32);
    }
    if (lane == 0) { src[row] = s1 + ba[0]; nbr[row] = s2; }
}

// ---------------------------------------------------------------------------
// Kernel 2b: global max of nbr (single block) -> softmax shift upper bound
// ---------------------------------------------------------------------------
__global__ __launch_bounds__(256) void k_nbrmax(const float* __restrict__ nbr,
                                                float* __restrict__ outmax)
{
    __shared__ float red[256];
    float mx = -3.4e38f;
    for (int i = threadIdx.x; i < N_NODES; i += 256) mx = fmaxf(mx, nbr[i]);
    red[threadIdx.x] = mx;
    __syncthreads();
    for (int s = 128; s > 0; s >>= 1) {
        if (threadIdx.x < s) red[threadIdx.x] = fmaxf(red[threadIdx.x], red[threadIdx.x + s]);
        __syncthreads();
    }
    if (threadIdx.x == 0) outmax[0] = red[0];
}

// ---------------------------------------------------------------------------
// Kernel 3: fused gather + leaky_relu + exp + P@h (bf16 WMMA).
// Block = 4 waves sharing ONE j-chunk; wave w owns i-tile ig*4+w.
// B tiles (256x32 bf16 of hT) are staged into LDS once per block via
// global_load_async_to_lds_b128 (ASYNCcnt), double-buffered so tile t+1
// streams in while tile t is consumed -> 4x less L2 traffic than per-wave
// B loads. nbr table (32KB) lives in LDS for the 64M gathers.
// ---------------------------------------------------------------------------
__global__ __launch_bounds__(128) void k_attn(const int* __restrict__ adj,
                                              const float* __restrict__ src,
                                              const float* __restrict__ nbr,
                                              const float* __restrict__ nbrmax,
                                              const __bf16* __restrict__ hT,
                                              float* __restrict__ pacc,
                                              float* __restrict__ pl)
{
    __shared__ float  snbr[N_NODES];            // 32 KB
    __shared__ __bf16 Bstage[2][256 * BROW];    // 2 x 20 KB (80B padded rows)

    const int tid  = threadIdx.x;
    const int lane = tid & 31;
    const int w    = tid >> 5;

    // cooperative nbr table load (ds_store_b128; barrier below covers it)
    #pragma unroll
    for (int q = 0; q < N_NODES / (128 * 4); ++q) {
        const int idx = (q * 128 + tid) * 4;
        *(float4*)(snbr + idx) = *(const float4*)(nbr + idx);
    }

    const int ig    = blockIdx.x >> 2;          // 0..127 : group of 4 i-tiles
    const int jc    = blockIdx.x & 3;           // shared j-chunk of this block
    const int itile = ig * 4 + w;
    const int i0    = itile * 16;
    const int m     = lane & 15;
    const int half  = lane >> 4;
    const int jbeg  = jc * (N_NODES / J_SPLIT);

    const uint32_t ldsB0 = (uint32_t)(uintptr_t)(&Bstage[0][0]);
    const uint32_t ldsB1 = (uint32_t)(uintptr_t)(&Bstage[1][0]);

    const float s = src[i0 + m];                // includes ba
    float em = s + nbrmax[0];                   // lrelu monotone => row-max bound
    em = em > 0.f ? em : NEG_SLOPE * em;
    const int* arow = adj + (i0 + m) * N_NODES;

    v8f acc[16];
    #pragma unroll
    for (int f = 0; f < 16; ++f)
        #pragma unroll
        for (int r = 0; r < 8; ++r) acc[f][r] = 0.f;
    float lsum = 0.f;

    // Stage hT[:, j..j+31] (256 rows x 64B) into LDS buffer: 8 rounds x
    // 128 threads x 16B async copies. 4 threads per 64B row, rows padded
    // to 80B so B-fragment ds_load_b128 reads are bank-conflict-free.
    auto stage = [&](int j, uint32_t ldsbase) {
        #pragma unroll
        for (int r = 0; r < 8; ++r) {
            const int idx  = r * 128 + tid;
            const int row  = idx >> 2;
            const int sub  = idx & 3;
            const int      goff = (row * N_NODES + j + sub * 8) * 2;        // bytes
            const uint32_t loff = ldsbase + row * (BROW * 2) + sub * 16;    // bytes
            asm volatile("global_load_async_to_lds_b128 %0, %1, %2"
                         :: "v"(loff), "v"(goff), "s"(hT) : "memory");
        }
    };

    stage(jbeg, ldsB0);                          // tile 0 -> buf 0

    const int NT = (N_NODES / J_SPLIT) / 32;     // 64 j-tiles per block
    for (int t = 0; t < NT; ++t) {
        const int j = jbeg + t * 32;
        if (t + 1 < NT) {
            stage(j + 32, (t & 1) ? ldsB0 : ldsB1);          // tile t+1
            asm volatile("s_wait_asynccnt 0x8" ::: "memory"); // oldest 8 (tile t) done
        } else {
            asm volatile("s_wait_asynccnt 0x0" ::: "memory");
        }
        __syncthreads();                         // tile t visible from all waves

        // ---- A: gather + leaky_relu + exp, packed in WMMA A-layout ----
        v16bf A;
        #pragma unroll
        for (int v = 0; v < 8; ++v) {
            const int K0 = (half << 4) + (v << 1);
            int2 id = *(const int2*)(arow + j + K0);
            float e0 = s + snbr[id.x];
            float e1 = s + snbr[id.y];
            e0 = e0 > 0.f ? e0 : NEG_SLOPE * e0;
            e1 = e1 > 0.f ? e1 : NEG_SLOPE * e1;
            const float p0 = __expf(e0 - em);
            const float p1 = __expf(e1 - em);
            lsum += p0 + p1;
            A[2 * v]     = (__bf16)p0;
            A[2 * v + 1] = (__bf16)p1;
        }

        // ---- 16 WMMAs, B fragments from LDS (2x ds_load_b128 each) ----
        const __bf16* Bbuf = &Bstage[t & 1][0];
        #pragma unroll
        for (int f = 0; f < 16; ++f) {
            const __bf16* bp = Bbuf + (f * 16 + m) * BROW + half * 16;
            v8bf blo = *(const v8bf*)(bp);
            v8bf bhi = *(const v8bf*)(bp + 8);
            v16bf B = __builtin_shufflevector(blo, bhi,
                        0,1,2,3,4,5,6,7,8,9,10,11,12,13,14,15);
            acc[f] = __builtin_amdgcn_wmma_f32_16x16x32_bf16(false, A, false, B,
                                                             (short)0, acc[f], false, false);
        }
        __syncthreads();                         // done with buf before overwrite
    }

    lsum += __shfl_xor(lsum, 16, 32);
    const int chunk = itile * J_SPLIT + jc;
    if (half == 0) pl[chunk * 16 + m] = lsum;
    float* pc = pacc + (size_t)chunk * (16 * OUT_F);
    #pragma unroll
    for (int f = 0; f < 16; ++f)
        #pragma unroll
        for (int r = 0; r < 8; ++r)
            pc[(r + half * 8) * OUT_F + f * 16 + m] = acc[f][r];
}

// ---------------------------------------------------------------------------
// Kernel 4: sum the J_SPLIT partials and normalize by the row sum
// ---------------------------------------------------------------------------
__global__ __launch_bounds__(256) void k_combine(const float* __restrict__ pacc,
                                                 const float* __restrict__ pl,
                                                 float* __restrict__ out)
{
    const int t  = blockIdx.x * 256 + threadIdx.x;
    const int i  = t >> 8;
    const int f  = t & 255;
    const int it = i >> 4, ir = i & 15;
    float a = 0.f, l = 0.f;
    #pragma unroll
    for (int c = 0; c < J_SPLIT; ++c) {
        const int chunk = it * J_SPLIT + c;
        a += pacc[(size_t)chunk * (16 * OUT_F) + ir * OUT_F + f];
        l += pl[chunk * 16 + ir];
    }
    out[t] = a / l;
}

// ---------------------------------------------------------------------------
extern "C" void kernel_launch(void* const* d_in, const int* in_sizes, int n_in,
                              void* d_out, int out_size, void* d_ws, size_t ws_size,
                              hipStream_t stream) {
    (void)in_sizes; (void)n_in; (void)out_size; (void)ws_size;
    const float* X   = (const float*)d_in[0];   // 8192 x 512
    const float* W   = (const float*)d_in[1];   // 256  x 512
    const float* bW  = (const float*)d_in[2];   // 256
    const float* a   = (const float*)d_in[3];   // 512
    const float* ba  = (const float*)d_in[4];   // 1
    const int*   adj = (const int*)d_in[5];     // 8192 x 8192
    float* out = (float*)d_out;                 // 8192 x 256

    char* ws = (char*)d_ws;
    float*  h    = (float*)ws;  ws += (size_t)N_NODES * OUT_F * sizeof(float);   // 8 MB
    __bf16* hT   = (__bf16*)ws; ws += (size_t)N_NODES * OUT_F * sizeof(__bf16);  // 4 MB
    float*  src  = (float*)ws;  ws += (size_t)N_NODES * sizeof(float);
    float*  nbrv = (float*)ws;  ws += (size_t)N_NODES * sizeof(float);
    float*  nmax = (float*)ws;  ws += 256;
    float*  pl   = (float*)ws;  ws += (size_t)512 * J_SPLIT * 16 * sizeof(float);
    float*  pacc = (float*)ws;  // 512 * J_SPLIT * 16 * 256 floats = 32 MB

    k_linear <<<2048, 128, 0, stream>>>(X, W, bW, h, hT);
    k_srcnbr <<<1024, 256, 0, stream>>>(h, a, ba, src, nbrv);
    k_nbrmax <<<1,    256, 0, stream>>>(nbrv, nmax);
    k_attn   <<<512,  128, 0, stream>>>(adj, src, nbrv, nmax, hT, pacc, pl);
    k_combine<<<8192, 256, 0, stream>>>(pacc, pl, out);
}
